// LinearAttention_53094385713649
// MI455X (gfx1250) — matrix operand order repaired and checked by reading, not compiled, not run on hardware
//
#include <hip/hip_runtime.h>
#include <math.h>

typedef __attribute__((ext_vector_type(16))) __bf16 v16bf;
typedef __attribute__((ext_vector_type(8)))  __bf16 v8bf;
typedef __attribute__((ext_vector_type(8)))  float  v8f;

static constexpr int BB = 4;
static constexpr int S  = 4096;
static constexpr int D  = 2048;
static constexpr int H  = 16;
static constexpr int DK = 128;
static constexpr int NR = BB * S;     // 16384 total rows
static constexpr int KT = D / 32;     // 64 k-tiles along the reduction dim
static constexpr int KS_CH = 8;       // S-chunks for the K_sum reduction

// ---------------------------------------------------------------------------
// fp32 -> bf16 convert (grid-stride)
// ---------------------------------------------------------------------------
__global__ void la_cvt(const float* __restrict__ in, __bf16* __restrict__ out, int n) {
    int i = blockIdx.x * blockDim.x + threadIdx.x;
    int stride = gridDim.x * blockDim.x;
    for (; i < n; i += stride) out[i] = (__bf16)in[i];
}

// ---------------------------------------------------------------------------
// Pack weight w[out][in] (fp32) into bf16 WMMA B-fragment layout:
//   packed[((tileN*KT + tileK)*32 + lane)*16 + i]
//     = w[tileN*16 + (lane&15)][tileK*32 + (i&7) + 8*(lane>>4) + 16*(i>>3)]
// so a B fragment is one contiguous 32-byte read per lane.
// ---------------------------------------------------------------------------
__global__ void la_pack_weight(const float* __restrict__ w, __bf16* __restrict__ out) {
    int idx = blockIdx.x * blockDim.x + threadIdx.x;        // 0 .. D*D-1
    int i    = idx & 15;
    int lane = (idx >> 4) & 31;
    int t    = idx >> 9;
    int tileK = t % KT;
    int tileN = t / KT;
    int o = tileN * 16 + (lane & 15);
    int k = tileK * 32 + (i & 7) + 8 * (lane >> 4) + 16 * (i >> 3);
    out[idx] = (__bf16)w[(size_t)o * D + k];
}

// ---------------------------------------------------------------------------
// Pack per-head KV (fp32, [bh][k=128][e=128]) into bf16 B-fragment layout:
//   packed[bh*16384 + ((t*4 + kt)*32 + lane)*16 + i]
//     = KV[bh][kt*32 + (i&7) + 8*(lane>>4) + 16*(i>>3)][t*16 + (lane&15)]
// ---------------------------------------------------------------------------
__global__ void la_pack_kv(const float* __restrict__ KV, __bf16* __restrict__ out) {
    int idx = blockIdx.x * blockDim.x + threadIdx.x;        // BB*H*DK*DK
    if (idx >= BB * H * DK * DK) return;
    int i    = idx & 15;
    int lane = (idx >> 4) & 31;
    int q    = idx >> 9;
    int kt = q & 3;          // DK/32 = 4 k-tiles
    int t  = (q >> 2) & 7;   // DK/16 = 8 n-tiles
    int bh = q >> 5;
    int k = kt * 32 + (i & 7) + 8 * (lane >> 4) + 16 * (i >> 3);
    int e = t * 16 + (lane & 15);
    out[idx] = (__bf16)KV[((size_t)bh * DK + k) * DK + e];
}

// ---------------------------------------------------------------------------
// Big GEMM: out[row, col] = sum_k A[row,k] * Bpacked  (+bias, opt. elu+1)
//  mode 0/1: feature map elu(x)+1, write per-head bf16 [b,h,s,dk]
//  mode 2  : no feature map,       write per-head bf16
//  mode 3  : no feature map,       write flat fp32 [row*D + col]
//
// grid = (NR/256, D/64), block = 256 (8 waves).
// Whole 64-col x 2048-K packed B panel (256 KB) staged into LDS once.
// Wave tile = 32(m) x 64(n).  The k-loop is unrolled x2 with ping-pong A
// register buffers (a <-> c) so the double-buffer needs no copies.
// ---------------------------------------------------------------------------
__global__ __launch_bounds__(256)
void la_gemm(const __bf16* __restrict__ A, const __bf16* __restrict__ Bp,
             const float* __restrict__ bias, int mode,
             __bf16* __restrict__ outH, float* __restrict__ outF) {
    extern __shared__ __bf16 sB[];                 // 4*KT*512 bf16 = 256 KB

    const int wave = threadIdx.x >> 5;
    const int lane = threadIdx.x & 31;
    const int r  = lane & 15;
    const int hf = lane >> 4;
    const int m0 = blockIdx.x * 256 + wave * 32;
    const int n0 = blockIdx.y * 64;

    // ---- stage the packed B panel (contiguous 256 KB region) into LDS ----
    const __bf16* bsrc = Bp + (size_t)(n0 >> 4) * KT * 512;
    for (int i = threadIdx.x * 8; i < 4 * KT * 512; i += 256 * 8)
        *(v8bf*)(sB + i) = *(const v8bf*)(bsrc + i);
    __syncthreads();

    union AB { v16bf v; v8bf h[2]; };

    const __bf16* arow0 = A + (size_t)(m0 + r) * D + 8 * hf;
    const __bf16* arow1 = arow0 + (size_t)16 * D;

    AB a0, a1, c0, c1;
    a0.h[0] = *(const v8bf*)(arow0);
    a0.h[1] = *(const v8bf*)(arow0 + 16);
    a1.h[0] = *(const v8bf*)(arow1);
    a1.h[1] = *(const v8bf*)(arow1 + 16);

    v8f acc[8] = {};

    for (int kb = 0; kb < D; kb += 64) {
        // preload k-step kb+32 into the c-buffers (always in range)
        __builtin_prefetch(arow0 + kb + 1024, 0, 3);
        c0.h[0] = *(const v8bf*)(arow0 + kb + 32);
        c0.h[1] = *(const v8bf*)(arow0 + kb + 48);
        c1.h[0] = *(const v8bf*)(arow1 + kb + 32);
        c1.h[1] = *(const v8bf*)(arow1 + kb + 48);
        {
            const __bf16* bbase = sB + (size_t)(kb >> 5) * 512 + lane * 16;
            v16bf bfr[4];
#pragma unroll
            for (int t = 0; t < 4; ++t)
                bfr[t] = *(const v16bf*)(bbase + (size_t)t * KT * 512);
#pragma unroll
            for (int t = 0; t < 4; ++t) {
                acc[t] = __builtin_amdgcn_wmma_f32_16x16x32_bf16(
                    false, a0.v, false, bfr[t], (short)0, acc[t], false, false);
                acc[4 + t] = __builtin_amdgcn_wmma_f32_16x16x32_bf16(
                    false, a1.v, false, bfr[t], (short)0, acc[4 + t], false, false);
            }
        }
        if (kb + 64 < D) {                          // preload kb+64 into a
            a0.h[0] = *(const v8bf*)(arow0 + kb + 64);
            a0.h[1] = *(const v8bf*)(arow0 + kb + 80);
            a1.h[0] = *(const v8bf*)(arow1 + kb + 64);
            a1.h[1] = *(const v8bf*)(arow1 + kb + 80);
        }
        {
            const __bf16* bbase = sB + (size_t)((kb + 32) >> 5) * 512 + lane * 16;
            v16bf bfr[4];
#pragma unroll
            for (int t = 0; t < 4; ++t)
                bfr[t] = *(const v16bf*)(bbase + (size_t)t * KT * 512);
#pragma unroll
            for (int t = 0; t < 4; ++t) {
                acc[t] = __builtin_amdgcn_wmma_f32_16x16x32_bf16(
                    false, c0.v, false, bfr[t], (short)0, acc[t], false, false);
                acc[4 + t] = __builtin_amdgcn_wmma_f32_16x16x32_bf16(
                    false, c1.v, false, bfr[t], (short)0, acc[4 + t], false, false);
            }
        }
    }

#pragma unroll
    for (int g = 0; g < 2; ++g) {
#pragma unroll
        for (int t = 0; t < 4; ++t) {
            const int col = n0 + t * 16 + r;
            const float bcol = bias[col];
#pragma unroll
            for (int j = 0; j < 8; ++j) {
                const int row = m0 + 16 * g + j + 8 * hf;
                float v = acc[4 * g + t][j] + bcol;
                if (mode <= 1) v = (v > 0.0f) ? (v + 1.0f) : __expf(v);
                if (mode <= 2) {
                    int b  = row >> 12;          // / S
                    int s  = row & (S - 1);
                    int hh = col >> 7;           // / DK
                    int dk = col & (DK - 1);
                    outH[((size_t)(b * H + hh) * S + s) * DK + dk] = (__bf16)v;
                } else {
                    outF[(size_t)row * D + col] = v;
                }
            }
        }
    }
}

// ---------------------------------------------------------------------------
// K_sum stage 1: partial[bh][c][sh][d] = sum over s-chunk (deterministic)
// grid = BB*H*KS_CH blocks x 256 threads; coalesced 256 B rows
// ---------------------------------------------------------------------------
__global__ void la_ksum_part(const __bf16* __restrict__ Kf, float* __restrict__ part) {
    int blk = blockIdx.x;
    int c  = blk % KS_CH;
    int bh = blk / KS_CH;
    int d  = threadIdx.x & (DK - 1);
    int sh = threadIdx.x >> 7;                      // 0/1
    const int chunk = S / KS_CH;                    // 512
    const __bf16* p = Kf + ((size_t)bh * S + c * chunk + sh) * DK + d;
    float s = 0.f;
    for (int i = 0; i < chunk; i += 2) s += (float)p[(size_t)i * DK];
    part[(((size_t)bh * KS_CH + c) * 2 + sh) * DK + d] = s;
}

// K_sum stage 2: combine the 2*KS_CH partials per (bh, d)
__global__ void la_ksum_final(const float* __restrict__ part, float* __restrict__ Ksum) {
    int idx = blockIdx.x * blockDim.x + threadIdx.x;        // BB*H*DK
    if (idx >= BB * H * DK) return;
    int bh = idx / DK, d = idx % DK;
    float s = 0.f;
#pragma unroll
    for (int j = 0; j < KS_CH * 2; ++j)
        s += part[((size_t)bh * KS_CH * 2 + j) * DK + d];
    Ksum[idx] = s;
}

// ---------------------------------------------------------------------------
// QK_sum[b,h,s] = dot(Q[b,h,s,:], K_sum[b,h,:])   (vectorized row loads)
// ---------------------------------------------------------------------------
__global__ void la_qksum(const __bf16* __restrict__ Qf, const float* __restrict__ Ksum,
                         float* __restrict__ QKsum) {
    int idx = blockIdx.x * blockDim.x + threadIdx.x;        // B*H*S
    if (idx >= BB * H * S) return;
    int bh = idx / S, s = idx % S;
    const v8bf* qv = (const v8bf*)(Qf + ((size_t)bh * S + s) * DK);
    const float* ks = Ksum + bh * DK;
    float acc = 0.f;
#pragma unroll
    for (int c = 0; c < DK / 8; ++c) {
        v8bf qq = qv[c];
#pragma unroll
        for (int j = 0; j < 8; ++j) acc += (float)qq[j] * ks[c * 8 + j];
    }
    QKsum[idx] = acc;
}

// ---------------------------------------------------------------------------
// KV[b,h,d,e] = sum_s K[b,h,s,d] * V[b,h,s,e]
// one block per (b,h).  Each 32-s step: cooperatively transpose the
// 32x128 K and V tiles into padded LDS (row stride 48 elems = 96 B, keeps
// 16-byte alignment), then build both WMMA fragments with ds_load_b128.
// ---------------------------------------------------------------------------
__global__ __launch_bounds__(256)
void la_kv(const __bf16* __restrict__ Kf, const __bf16* __restrict__ Vf,
           float* __restrict__ KV) {
    __shared__ __bf16 sKT[DK][48];   // [d][s_local], 12 KB
    __shared__ __bf16 sVT[DK][48];   // [e][s_local], 12 KB

    const int bh   = blockIdx.x;
    const int wave = threadIdx.x >> 5;
    const int lane = threadIdx.x & 31;
    const int r  = lane & 15;
    const int hf = lane >> 4;
    const int m0 = wave * 16;

    const __bf16* Kb = Kf + (size_t)bh * S * DK;
    const __bf16* Vb = Vf + (size_t)bh * S * DK;

    v8f acc[8] = {};

    const int sl = threadIdx.x >> 3;          // 0..31  (s within tile)
    const int dg = (threadIdx.x & 7) * 16;    // 0,16,...,112

    for (int s0 = 0; s0 < S; s0 += 32) {
        __syncthreads();                       // protect prior fragment reads
        const __bf16* kp = Kb + (size_t)(s0 + sl) * DK + dg;
        const __bf16* vp = Vb + (size_t)(s0 + sl) * DK + dg;
#pragma unroll
        for (int q = 0; q < 16; ++q) {
            sKT[dg + q][sl] = kp[q];
            sVT[dg + q][sl] = vp[q];
        }
        __syncthreads();

        union AB { v16bf v; v8bf h[2]; } a;
        a.h[0] = *(const v8bf*)(&sKT[m0 + r][8 * hf]);
        a.h[1] = *(const v8bf*)(&sKT[m0 + r][16 + 8 * hf]);
#pragma unroll
        for (int t = 0; t < 8; ++t) {
            union AB b;
            b.h[0] = *(const v8bf*)(&sVT[t * 16 + r][8 * hf]);
            b.h[1] = *(const v8bf*)(&sVT[t * 16 + r][16 + 8 * hf]);
            acc[t] = __builtin_amdgcn_wmma_f32_16x16x32_bf16(
                false, a.v, false, b.v, (short)0, acc[t], false, false);
        }
    }

    float* KVb = KV + (size_t)bh * DK * DK;
#pragma unroll
    for (int t = 0; t < 8; ++t) {
#pragma unroll
        for (int j = 0; j < 8; ++j) {
            int row = m0 + j + 8 * hf;
            int e   = t * 16 + r;
            KVb[(size_t)row * DK + e] = acc[t][j];
        }
    }
}

// ---------------------------------------------------------------------------
// out[b,h,s,e] = (Q[b,h,s,:] @ KVp[b,h,:,e]) / (QK_sum[b,h,s] + 1e-8)
// KVp is pre-packed bf16 B-fragments -> one 32-byte load per fragment.
// grid = (B*H, S/128); wave -> 16 rows x 128 cols, K = 128 (4 WMMA steps)
// ---------------------------------------------------------------------------
__global__ __launch_bounds__(256)
void la_qkv(const __bf16* __restrict__ Qf, const __bf16* __restrict__ KVp,
            const float* __restrict__ QKsum, __bf16* __restrict__ attn) {
    const int bh   = blockIdx.x;
    const int wave = threadIdx.x >> 5;
    const int lane = threadIdx.x & 31;
    const int r  = lane & 15;
    const int hf = lane >> 4;
    const int s0 = blockIdx.y * 128 + wave * 16;

    const __bf16* Qb  = Qf + (size_t)bh * S * DK;
    const __bf16* KVb = KVp + (size_t)bh * DK * DK;   // packed fragments

    v8f acc[8] = {};
    const __bf16* qrow = Qb + (size_t)(s0 + r) * DK;

#pragma unroll
    for (int kb = 0; kb < DK; kb += 32) {
        union AB { v16bf v; v8bf h[2]; } a;
        a.h[0] = *(const v8bf*)(qrow + kb + 8 * hf);
        a.h[1] = *(const v8bf*)(qrow + kb + 16 + 8 * hf);
        const int kt = kb >> 5;
#pragma unroll
        for (int t = 0; t < 8; ++t) {
            v16bf b = *(const v16bf*)(KVb + (((size_t)t * 4 + kt) * 32 + lane) * 16);
            acc[t] = __builtin_amdgcn_wmma_f32_16x16x32_bf16(
                false, a.v, false, b, (short)0, acc[t], false, false);
        }
    }

    const int bidx = bh / H, hh = bh % H;
#pragma unroll
    for (int t = 0; t < 8; ++t) {
        int e = t * 16 + r;
#pragma unroll
        for (int j = 0; j < 8; ++j) {
            int s = s0 + j + 8 * hf;
            float denom = QKsum[(size_t)bh * S + s] + 1e-8f;
            float v = acc[t][j] / denom;
            attn[(size_t)(bidx * S + s) * D + hh * DK + e] = (__bf16)v;
        }
    }
}

// ---------------------------------------------------------------------------
extern "C" void kernel_launch(void* const* d_in, const int* in_sizes, int n_in,
                              void* d_out, int out_size, void* d_ws, size_t ws_size,
                              hipStream_t stream) {
    const float* x  = (const float*)d_in[0];
    const float* wq = (const float*)d_in[1];
    const float* bq = (const float*)d_in[2];
    const float* wk = (const float*)d_in[3];
    const float* bk = (const float*)d_in[4];
    const float* wv = (const float*)d_in[5];
    const float* bv = (const float*)d_in[6];
    const float* wo = (const float*)d_in[7];
    const float* bo = (const float*)d_in[8];

    char* ws = (char*)d_ws;
    size_t off = 0;
    auto take = [&](size_t bytes) { char* p = ws + off; off += (bytes + 255) & ~size_t(255); return p; };

    __bf16* xb    = (__bf16*)take((size_t)NR * D * 2);   // also reused as attn context
    __bf16* wq_p  = (__bf16*)take((size_t)D * D * 2);
    __bf16* wk_p  = (__bf16*)take((size_t)D * D * 2);
    __bf16* wv_p  = (__bf16*)take((size_t)D * D * 2);
    __bf16* wo_p  = (__bf16*)take((size_t)D * D * 2);
    __bf16* Qf    = (__bf16*)take((size_t)BB * H * S * DK * 2);
    __bf16* Kf    = (__bf16*)take((size_t)BB * H * S * DK * 2);
    __bf16* Vf    = (__bf16*)take((size_t)BB * H * S * DK * 2);
    float*  KV    = (float*)take((size_t)BB * H * DK * DK * 4);
    __bf16* KVp   = (__bf16*)take((size_t)BB * H * DK * DK * 2);
    float*  Kpart = (float*)take((size_t)BB * H * KS_CH * 2 * DK * 4);
    float*  Ksum  = (float*)take((size_t)BB * H * DK * 4);
    float*  QKs   = (float*)take((size_t)BB * H * S * 4);

    // 1. convert x to bf16
    la_cvt<<<4096, 256, 0, stream>>>(x, xb, NR * D);

    // 2. pack weights into WMMA fragment layout
    la_pack_weight<<<(D * D) / 256, 256, 0, stream>>>(wq, wq_p);
    la_pack_weight<<<(D * D) / 256, 256, 0, stream>>>(wk, wk_p);
    la_pack_weight<<<(D * D) / 256, 256, 0, stream>>>(wv, wv_p);
    la_pack_weight<<<(D * D) / 256, 256, 0, stream>>>(wo, wo_p);

    // 3. Q/K/V projections (feature map fused for Q,K); 256 KB dynamic LDS
    dim3 gg(NR / 256, D / 64);
    const size_t lds = (size_t)4 * KT * 512 * sizeof(__bf16);   // 256 KB
    la_gemm<<<gg, 256, lds, stream>>>(xb, wq_p, bq, 0, Qf, nullptr);
    la_gemm<<<gg, 256, lds, stream>>>(xb, wk_p, bk, 1, Kf, nullptr);
    la_gemm<<<gg, 256, lds, stream>>>(xb, wv_p, bv, 2, Vf, nullptr);

    // 4. K_sum (two-stage, deterministic)
    la_ksum_part<<<BB * H * KS_CH, 256, 0, stream>>>(Kf, Kpart);
    la_ksum_final<<<(BB * H * DK + 255) / 256, 256, 0, stream>>>(Kpart, Ksum);

    // 5. KV = K^T V per head (LDS-transposed tiles)
    la_kv<<<BB * H, 256, 0, stream>>>(Kf, Vf, KV);

    // 5b. pack KV into bf16 B-fragments
    la_pack_kv<<<(BB * H * DK * DK + 255) / 256, 256, 0, stream>>>(KV, KVp);

    // 6. QK_sum
    la_qksum<<<(BB * H * S + 255) / 256, 256, 0, stream>>>(Qf, Ksum, QKs);

    // 7. context = (Q @ KV) / (QK_sum + eps)  -> overwrite xb (x no longer needed)
    dim3 gq(BB * H, S / 128);
    la_qkv<<<gq, 256, 0, stream>>>(Qf, KVp, QKs, xb);

    // 8. final projection -> fp32 output
    la_gemm<<<gg, 256, lds, stream>>>(xb, wo_p, bo, 3, nullptr, (float*)d_out);
}
